// AdaptiveFeatureSelector_52682068853139
// MI455X (gfx1250) — matrix operand order repaired and checked by reading, not compiled, not run on hardware
//
#include <hip/hip_runtime.h>
#include <hip/hip_bf16.h>

// ---------------------------------------------------------------------------
// AdaptiveFeatureSelector fused kernel for gfx1250 (CDNA5, wave32, WMMA bf16)
//
// Shapes: B=65536, D=512, H=128, K=358 (output cols, padded to 368 for WMMA)
// All GEMMs via v_wmma_f32_16x16x32_bf16. One wave = one 16-row tile.
// ---------------------------------------------------------------------------

typedef __attribute__((ext_vector_type(16))) __bf16 v16bf;
typedef __attribute__((ext_vector_type(8)))  float  v8f;

#define D_IN   512
#define H_DIM  128
#define K_OUT  358
#define NPAD   368
#define BATCH  65536

// LDS row strides (elements); multiples of 8 so 16B-aligned vector LDS ops work
#define SX 520   // x / combined row stride (512 + 8 pad)
#define SH 136   // hidden row stride (128 + 8 pad)

// packed-weight offsets in d_ws (bf16 elements)
#define OFF_W1   0
#define OFF_W2   65536
#define OFF_W3   81920
#define OFF_WG1  147456
#define OFF_WG2  212992
#define OFF_WR1  278528
#define OFF_WR2  344064
// total = 391168 bf16 = 782336 bytes

// ---------------------------------------------------------------------------
// Weight prepack: f32 row-major [Ktot x Nsrc]  ->  bf16 WMMA B-fragment order.
// Fragment f = nt*(Ktot/32) + kt holds the 32x16 tile (K=kt*32.., N=nt*16..).
// Within a fragment (512 bf16): lane l, half j:
//   column n = nt*16 + (l & 15)
//   k        = kt*32 + (l >= 16 ? 16 : 0) + j          (j = 0..15)
// so each lane's 16 halves are contiguous: packed[f*512 + l*16 + j].
// ---------------------------------------------------------------------------
__global__ __launch_bounds__(256) void afs_prepack(
    const float* __restrict__ src, __bf16* __restrict__ dst,
    int Ktot, int Nsrc, int Npad)
{
    const int idx   = blockIdx.x * 256 + threadIdx.x;
    const int total = (Ktot / 32) * (Npad / 16) * 512;
    if (idx >= total) return;
    const int f  = idx >> 9;
    const int e  = idx & 511;
    const int l  = e >> 4;
    const int j  = e & 15;
    const int kt = f % (Ktot / 32);
    const int nt = f / (Ktot / 32);
    const int n  = nt * 16 + (l & 15);
    const int k  = kt * 32 + ((l >> 4) << 4) + j;
    const float v = (n < Nsrc) ? src[(size_t)k * Nsrc + n] : 0.0f;
    dst[idx] = (__bf16)v;
}

// ---------------------------------------------------------------------------
// Fragment loaders
// ---------------------------------------------------------------------------
// A (16x32 bf16, M x K) from LDS row-major buffer:
//   lane l (0..15): row m=l,   halves = K { 0..15 }
//   lane l (16..31): row m=l-16, halves = K { 16..31 }
__device__ __forceinline__ v16bf load_A_frag(const __bf16* base, int stride, int kt)
{
    const int l  = threadIdx.x & 31;
    const int m  = l & 15;
    const int hi = l >> 4;
    const __bf16* rp = base + m * stride + kt * 32 + hi * 16;
    union { v16bf v; uint4 q[2]; } u;
    u.q[0] = *(const uint4*)(rp);        // K halves  [hi*16 .. hi*16+7]
    u.q[1] = *(const uint4*)(rp + 8);    // K halves  [hi*16+8 .. hi*16+15]
    return u.v;
}

// B (32x16 bf16, K x N) from packed global weights: lane-contiguous 32 bytes.
__device__ __forceinline__ v16bf load_B_frag(const __bf16* wpk, int frag)
{
    const int l = threadIdx.x & 31;
    const __bf16* p = wpk + frag * 512 + l * 16;
    union { v16bf v; uint4 q[2]; } u;
    u.q[0] = *(const uint4*)(p);
    u.q[1] = *(const uint4*)(p + 8);
    return u.v;
}

__device__ __forceinline__ v8f wmma_bf16(v16bf a, v16bf b, v8f c)
{
    return __builtin_amdgcn_wmma_f32_16x16x32_bf16(
        /*neg_a=*/false, a, /*neg_b=*/false, b,
        /*c_mod=*/(short)0, c, /*reuse_a=*/false, /*reuse_b=*/false);
}

// ---------------------------------------------------------------------------
// One wave computes act(A[16 x 32*KTILES] @ W + bias) for NTILES column tiles.
// MODE 0: relu    -> bf16 LDS      MODE 1: sigmoid -> f32 LDS
// MODE 2: sigmoid *= f32 LDS       MODE 3: identity -> global (col < ncols)
// D-fragment layout: VGPR r holds element (M = r + 8*(lane>=16), N = lane&15).
// ---------------------------------------------------------------------------
template <int KTILES, int NTILES, int MODE>
__device__ __forceinline__ void wave_gemm(
    const __bf16* __restrict__ A, int strideA,
    const __bf16* __restrict__ Wpk, const float* __restrict__ bias,
    __bf16* OutB, int strideOB,
    float*  OutF, int strideOF,
    float* __restrict__ gout, int rowBase, int ncols)
{
    const int l  = threadIdx.x & 31;
    const int nn = l & 15;
    const int hi = l >> 4;

    for (int nt = 0; nt < NTILES; ++nt) {
        const __bf16* wp = Wpk + (size_t)nt * KTILES * 512;
        __builtin_prefetch(wp + KTILES * 512, 0, 0);   // next column tile
        v8f acc = {};
#pragma unroll
        for (int kt = 0; kt < KTILES; ++kt) {
            v16bf a = load_A_frag(A, strideA, kt);
            v16bf b = load_B_frag(wp, kt);
            acc = wmma_bf16(a, b, acc);
        }
        const int col = nt * 16 + nn;
        const float bv = (MODE == 3) ? ((col < ncols) ? bias[col] : 0.0f)
                                     : bias[col];
#pragma unroll
        for (int r = 0; r < 8; ++r) {
            const int row = r + 8 * hi;
            const float v = acc[r] + bv;
            if (MODE == 0) {
                OutB[row * strideOB + col] = (__bf16)fmaxf(v, 0.0f);
            } else if (MODE == 1) {
                OutF[row * strideOF + col] = 1.0f / (1.0f + __expf(-v));
            } else if (MODE == 2) {
                OutF[row * strideOF + col] *= 1.0f / (1.0f + __expf(-v));
            } else {
                if (col < ncols)
                    gout[(size_t)(rowBase + row) * ncols + col] = v;
            }
        }
    }
}

// ---------------------------------------------------------------------------
// Fused main kernel: one wave per 16-row tile.
// ---------------------------------------------------------------------------
__global__ __launch_bounds__(32) void afs_main(
    const float*  __restrict__ x,
    const __bf16* __restrict__ w1,  const float* __restrict__ b1,
    const __bf16* __restrict__ w2,  const float* __restrict__ b2,
    const __bf16* __restrict__ w3,  const float* __restrict__ b3,
    const __bf16* __restrict__ wg1, const float* __restrict__ bg1,
    const __bf16* __restrict__ wg2, const float* __restrict__ bg2,
    const __bf16* __restrict__ wr1, const float* __restrict__ br1,
    const __bf16* __restrict__ wr2, const float* __restrict__ br2,
    float* __restrict__ out)
{
    __shared__ __bf16 xb[16 * SX];    // x tile (bf16), later masked in place
    __shared__ __bf16 hb[16 * SH];    // hidden buffer A
    __shared__ __bf16 hb2[16 * SH];   // hidden buffer B
    __shared__ float  cb[16 * SX];    // combined = sigmoid(imp)*sigmoid(gate)

    const int lane    = threadIdx.x & 31;
    const int rowBase = blockIdx.x * 16;

    // load + convert x tile to bf16 in LDS (coalesced)
    for (int i = lane; i < 16 * D_IN; i += 32) {
        const int m = i >> 9, c = i & 511;
        xb[m * SX + c] = (__bf16)x[(size_t)(rowBase + m) * D_IN + c];
    }
    // single-wave workgroup: LDS ops are in-order within a wave, no barrier.

    // importance net: D->H->H->D, sigmoid into cb
    wave_gemm<16, 8, 0>(xb,  SX, w1, b1,  hb,  SH, nullptr, 0, nullptr, 0, 0);
    wave_gemm< 4, 8, 0>(hb,  SH, w2, b2,  hb2, SH, nullptr, 0, nullptr, 0, 0);
    wave_gemm< 4, 32, 1>(hb2, SH, w3, b3,  nullptr, 0, cb, SX, nullptr, 0, 0);
    // gate net: D->H->D, sigmoid multiplied into cb
    wave_gemm<16, 8, 0>(xb,  SX, wg1, bg1, hb,  SH, nullptr, 0, nullptr, 0, 0);
    wave_gemm< 4, 32, 2>(hb,  SH, wg2, bg2, nullptr, 0, cb, SX, nullptr, 0, 0);

    // exact top-K (K=358 of 512) per row: scores are positive floats, so
    // uint bit order == float order. Radix-select the K-th largest value
    // bit-by-bit; whole wave cooperates on one row at a time.
    for (int m = 0; m < 16; ++m) {
        unsigned v[16];
#pragma unroll
        for (int j = 0; j < 16; ++j)
            v[j] = __float_as_uint(cb[m * SX + lane * 16 + j]);

        unsigned ans = 0u;
        for (int bit = 30; bit >= 0; --bit) {
            const unsigned cand = ans | (1u << bit);
            int c = 0;
#pragma unroll
            for (int j = 0; j < 16; ++j) c += (v[j] >= cand) ? 1 : 0;
            c += __shfl_xor(c, 16);
            c += __shfl_xor(c, 8);
            c += __shfl_xor(c, 4);
            c += __shfl_xor(c, 2);
            c += __shfl_xor(c, 1);
            if (c >= K_OUT) ans = cand;   // largest T with count(v>=T) >= K
        }
        // mask: zero x where score below the K-th largest
#pragma unroll
        for (int j = 0; j < 16; ++j)
            if (v[j] < ans) xb[m * SX + lane * 16 + j] = (__bf16)0.0f;
    }

    // reconstruction net: masked-x D->H->K(358)
    wave_gemm<16, 8, 0>(xb, SX, wr1, br1, hb, SH, nullptr, 0, nullptr, 0, 0);
    wave_gemm< 4, 23, 3>(hb, SH, wr2, br2, nullptr, 0, nullptr, 0,
                         out, rowBase, K_OUT);
}

// ---------------------------------------------------------------------------
// Host launcher
// ---------------------------------------------------------------------------
extern "C" void kernel_launch(void* const* d_in, const int* in_sizes, int n_in,
                              void* d_out, int out_size, void* d_ws, size_t ws_size,
                              hipStream_t stream)
{
    const float* x   = (const float*)d_in[0];
    const float* W1  = (const float*)d_in[1];
    const float* b1  = (const float*)d_in[2];
    const float* W2  = (const float*)d_in[3];
    const float* b2  = (const float*)d_in[4];
    const float* W3  = (const float*)d_in[5];
    const float* b3  = (const float*)d_in[6];
    const float* Wg1 = (const float*)d_in[7];
    const float* bg1 = (const float*)d_in[8];
    const float* Wg2 = (const float*)d_in[9];
    const float* bg2 = (const float*)d_in[10];
    const float* Wr1 = (const float*)d_in[11];
    const float* br1 = (const float*)d_in[12];
    const float* Wr2 = (const float*)d_in[13];
    const float* br2 = (const float*)d_in[14];
    float* out = (float*)d_out;

    __bf16* ws = (__bf16*)d_ws;
    __bf16* w1p  = ws + OFF_W1;
    __bf16* w2p  = ws + OFF_W2;
    __bf16* w3p  = ws + OFF_W3;
    __bf16* wg1p = ws + OFF_WG1;
    __bf16* wg2p = ws + OFF_WG2;
    __bf16* wr1p = ws + OFF_WR1;
    __bf16* wr2p = ws + OFF_WR2;

    auto pk = [&](const float* src, __bf16* dst, int Ktot, int Nsrc, int Npad) {
        const int total = (Ktot / 32) * (Npad / 16) * 512;
        afs_prepack<<<(total + 255) / 256, 256, 0, stream>>>(src, dst, Ktot, Nsrc, Npad);
    };
    pk(W1,  w1p,  D_IN,  H_DIM, H_DIM);   // 512x128
    pk(W2,  w2p,  H_DIM, H_DIM, H_DIM);   // 128x128
    pk(W3,  w3p,  H_DIM, D_IN,  D_IN);    // 128x512
    pk(Wg1, wg1p, D_IN,  H_DIM, H_DIM);   // 512x128
    pk(Wg2, wg2p, H_DIM, D_IN,  D_IN);    // 128x512
    pk(Wr1, wr1p, D_IN,  H_DIM, H_DIM);   // 512x128
    pk(Wr2, wr2p, H_DIM, K_OUT, NPAD);    // 128x358 -> padded 128x368

    afs_main<<<BATCH / 16, 32, 0, stream>>>(
        x, w1p, b1, w2p, b2, w3p, b3,
        wg1p, bg1, wg2p, bg2, wr1p, br1, wr2p, br2, out);
}
